// SkipLSTM_12945031430733
// MI455X (gfx1250) — compile-verified, matrix operand order checked by reading, not run on hardware
//
#include <hip/hip_runtime.h>
#include <hip/hip_bf16.h>
#include <cstdint>
#include <cstddef>

// ---------------- problem constants ----------------
#define Bx   64
#define Sx   512
#define INx  3
#define Hx   400
#define Hp   416          // H padded to 13*32 for WMMA K-tiling
#define G4   1600         // 4*H gate width
#define OUTx 121
#define OUTp 128
#define WINx 10
#define Ux   64
#define EMBx 64
#define KT   (Hp/32)      // 13 K-tiles of 32
#define NWG  25           // Hx/16 column-owning workgroups

typedef _Float16 half_t;
typedef __attribute__((ext_vector_type(16))) _Float16 v16h;
typedef __attribute__((ext_vector_type(8)))  float    v8f;

// ---------------- WMMA helpers ----------------
// Fragment convention (compile-only; layout stated, not runtime-verified):
// A (16x32 f16): lane l -> row (row0 + l%16), 16 contiguous halves starting at
//   k0 + 16*(l/16).  B (32x16 f16, stored as N rows x K cols = W row-major):
//   same addressing with row = n.  Both are 32B-aligned loads (Hp*2 = 832 = 26*32).
__device__ __forceinline__ v16h load_frag(const half_t* base, int pitch,
                                          int row0, int k0, int lane) {
  const half_t* p = base + (size_t)(row0 + (lane & 15)) * pitch
                         + k0 + ((lane >> 4) << 4);
  return *(const v16h*)p;
}

__device__ __forceinline__ v8f wmma16(v16h a, v16h b, v8f c) {
  return __builtin_amdgcn_wmma_f32_16x16x32_f16(
      false, a, false, b, (short)0, c, false, false);
}

__device__ __forceinline__ float sigf(float x) {
  return 1.0f / (1.0f + __expf(-x));
}

// ---------------- grid-wide sync (persistent kernels) ----------------
__device__ __forceinline__ void gsync(unsigned* ctr, unsigned target) {
  __threadfence();
  __syncthreads();
  if (threadIdx.x == 0) {
    atomicAdd(ctr, 1u);
    while (atomicAdd(ctr, 0u) < target) __builtin_amdgcn_s_sleep(2);
  }
  __syncthreads();
  __threadfence();
}

// ---------------- weight prep ----------------
__global__ void cvt_pad_kernel(const float* __restrict__ src, half_t* __restrict__ dst,
                               int rows, int cols, int pitch) {
  int i = blockIdx.x * blockDim.x + threadIdx.x;
  if (i < rows * cols) {
    int r = i / cols, c = i - r * cols;
    dst[(size_t)r * pitch + c] = (half_t)src[i];
  }
}

__global__ void bsum_kernel(const float* bih0, const float* bhh0,
                            const float* bihr, const float* bhhr, float* bsum) {
  int i = blockIdx.x * blockDim.x + threadIdx.x;
  if (i < 3 * G4) {
    int l = i / G4, j = i - l * G4;
    bsum[i] = (l == 0) ? (bih0[j] + bhh0[j])
                       : (bihr[(l - 1) * G4 + j] + bhhr[(l - 1) * G4 + j]);
  }
}

// Copy this WG's 64 weight rows (gate strips i,f,g,o at column j0) into LDS.
// LDS row index r = g*16 + rr  <->  global gate row g*Hx + j0 + rr.
__device__ __forceinline__ void cache_wslice(half_t* dstS, const half_t* W,
                                             int j0, int tid) {
  const int rowU4 = Hp / 8;  // 52 uint4 per row (Hp halves * 2B / 16B)
  for (int e = tid; e < 64 * rowU4; e += 128) {
    int r = e / rowU4, q = e - r * rowU4;
    int g = r >> 4, rr = r & 15;
    const uint4* src = (const uint4*)(W + (size_t)(g * Hx + j0 + rr) * Hp) + q;
    ((uint4*)(dstS + (size_t)r * Hp))[q] = *src;
  }
}

// ---------------- layer 0: LSTM + Gaussian window attention ----------------
__global__ __launch_bounds__(128)
void lstm0_kernel(const float* __restrict__ x, const float* __restrict__ seqs,
                  const float* __restrict__ ipW0, const float* __restrict__ ipb0,
                  const float* __restrict__ winW, const float* __restrict__ winb,
                  const float* __restrict__ whpW,
                  const half_t* __restrict__ Wih16, const half_t* __restrict__ Whh16,
                  const float* __restrict__ bsum,
                  half_t* hbufA, half_t* hbufB, half_t* wbufA, half_t* wbufB,
                  float* keta, half_t* hseq, half_t* wseq, unsigned* ctr) {
  // 3 x 53,248 B = 159,744 B static LDS: cached Whh slice, cached Wih slice,
  // per-step combined input tile (aliased by attention scratch in WG0).
  __shared__ __align__(16) unsigned char smem[3 * Bx * Hp * sizeof(half_t)];
  half_t* whhS = (half_t*)smem;
  half_t* wihS = whhS + Bx * Hp;
  half_t* inpS = wihS + Bx * Hp;

  const int tid = threadIdx.x;
  const int wave = tid >> 5, lane = tid & 31;
  const int jt = blockIdx.x, j0 = jt * 16;
  const int mbase = wave * 16;
  const int jlane = j0 + (lane & 15);

  // one-time: weight slices into LDS (stay resident for all 512 steps)
  cache_wslice(whhS, Whh16, j0, tid);
  cache_wslice(wihS, Wih16, j0, tid);

  // per-lane gate biases (column-fixed, hoisted for all 512 steps)
  const float bI = bsum[0 * Hx + jlane];
  const float bF = bsum[1 * Hx + jlane];
  const float bG = bsum[2 * Hx + jlane];
  const float bO = bsum[3 * Hx + jlane];

  float c[8];
#pragma unroll
  for (int r = 0; r < 8; r++) c[r] = 0.f;

  unsigned tgt = 0;
  for (int t = 0; t < Sx; t++) {
    half_t* hprev = (t & 1) ? hbufB : hbufA;
    half_t* hnew  = (t & 1) ? hbufA : hbufB;
    const half_t* wprev = (t & 1) ? wbufB : wbufA;
    half_t* wnew  = (t & 1) ? wbufA : wbufB;

    __syncthreads();
    // combined input tile: (x_t @ ipW0^T + ipb0 + w(t-1)) — both hit Wih,
    // so fold them and run ONE K=416 GEMM pass instead of two.
    for (int e = tid; e < Bx * Hx; e += 128) {
      int b = e / Hx, j = e - b * Hx;
      const float* xp = x + ((size_t)b * Sx + t) * INx;
      float v = ipb0[j] + xp[0] * ipW0[j * INx + 0]
                        + xp[1] * ipW0[j * INx + 1]
                        + xp[2] * ipW0[j * INx + 2]
                        + (float)wprev[(size_t)b * Hp + j];
      inpS[b * Hp + j] = (half_t)v;
    }
    for (int e = tid; e < Bx * (Hp - Hx); e += 128) {
      int b = e / (Hp - Hx), j = Hx + (e - b * (Hp - Hx));
      inpS[b * Hp + j] = (half_t)0.f;
    }
    __syncthreads();

    // gates = h@Whh^T + (inp+w)@Wih^T  (four 16-col strips i,f,g,o)
    v8f acc[4];
    {
      v8f z = {};
#pragma unroll
      for (int g = 0; g < 4; g++) acc[g] = z;
    }
#pragma unroll 1
    for (int kt = 0; kt < KT; kt++) {
      v16h a = load_frag(hprev, Hp, mbase, kt * 32, lane);
#pragma unroll
      for (int g = 0; g < 4; g++) {
        v16h bf = load_frag(whhS, Hp, g * 16, kt * 32, lane);
        acc[g] = wmma16(a, bf, acc[g]);
      }
    }
#pragma unroll 1
    for (int kt = 0; kt < KT; kt++) {
      v16h a = load_frag(inpS, Hp, mbase, kt * 32, lane);
#pragma unroll
      for (int g = 0; g < 4; g++) {
        v16h bf = load_frag(wihS, Hp, g * 16, kt * 32, lane);
        acc[g] = wmma16(a, bf, acc[g]);
      }
    }

    // LSTM cell (pure VALU; c lives in VGPRs for all 512 steps)
#pragma unroll
    for (int r = 0; r < 8; r++) {
      float gi = acc[0][r] + bI;
      float gf = acc[1][r] + bF;
      float gg = acc[2][r] + bG;
      float go = acc[3][r] + bO;
      float cn = sigf(gf) * c[r] + sigf(gi) * tanhf(gg);
      c[r] = cn;
      float hv = sigf(go) * tanhf(cn);
      int b = mbase + r + ((lane & 16) ? 8 : 0);
      half_t h16 = (half_t)hv;
      hnew[(size_t)b * Hp + jlane] = h16;
      hseq[((size_t)b * Sx + t) * Hp + jlane] = h16;
    }

    gsync(ctr, tgt += NWG);  // full h(t) visible

    if (jt == 0) {
      // Gaussian-mixture window attention; scratch aliases ONLY inpS
      float* aS = (float*)inpS;
      float* bS = aS + Bx * 16;
      float* kS = bS + Bx * 16;
      float* phiS = kS + Bx * 16;
      float* wvS = phiS + Bx * Ux;

      for (int e = tid; e < Bx * 3 * WINx; e += 128) {
        int b = e / (3 * WINx), k = e - b * 3 * WINx;
        float s = winb[k];
        const half_t* hp = hnew + (size_t)b * Hp;
        for (int j = 0; j < Hx; j++) s += (float)hp[j] * winW[k * Hx + j];
        if (k < WINx) aS[b * 16 + k] = s;
        else if (k < 2 * WINx) bS[b * 16 + (k - WINx)] = __expf(s);
        else {
          int kk = k - 2 * WINx;
          float kv = keta[b * WINx + kk] + __expf(s);
          keta[b * WINx + kk] = kv;
          kS[b * 16 + kk] = kv;
        }
      }
      __syncthreads();
      for (int e = tid; e < Bx * Ux; e += 128) {
        int b = e / Ux, u = e - b * Ux;
        float uu = (float)(u + 1);
        float s = 0.f;
#pragma unroll
        for (int k = 0; k < WINx; k++) {
          float d = kS[b * 16 + k] - uu;
          s += __expf(aS[b * 16 + k] - bS[b * 16 + k] * d * d);
        }
        phiS[b * Ux + u] = s;
      }
      __syncthreads();
      for (int e = tid; e < Bx * EMBx; e += 128) {
        int b = e / EMBx, em = e - b * EMBx;
        float s = 0.f;
        for (int u = 0; u < Ux; u++)
          s += phiS[b * Ux + u] * seqs[((size_t)b * Ux + u) * EMBx + em];
        wvS[b * EMBx + em] = s;
      }
      __syncthreads();
      for (int e = tid; e < Bx * Hx; e += 128) {
        int b = e / Hx, j = e - b * Hx;
        float s = 0.f;
        for (int em = 0; em < EMBx; em++)
          s += wvS[b * EMBx + em] * whpW[j * EMBx + em];
        half_t v = (half_t)s;
        wnew[(size_t)b * Hp + j] = v;
        wseq[((size_t)b * Sx + t) * Hp + j] = v;
      }
    }
    gsync(ctr, tgt += NWG);  // w(t+1 input) visible
  }
}

// ---------------- layers 1..2: plain LSTM with skip inputs ----------------
__global__ __launch_bounds__(128)
void lstmN_kernel(const float* __restrict__ x,
                  const float* __restrict__ ipWl, const float* __restrict__ ipbl,
                  const half_t* __restrict__ Wih16, const half_t* __restrict__ Whh16,
                  const float* __restrict__ bsum,
                  const half_t* __restrict__ hprev_seq, const half_t* __restrict__ wseq,
                  half_t* hbufA, half_t* hbufB, half_t* hseq, unsigned* ctr) {
  __shared__ __align__(16) unsigned char smem[3 * Bx * Hp * sizeof(half_t)];
  half_t* whhS = (half_t*)smem;
  half_t* wihS = whhS + Bx * Hp;
  half_t* inpS = wihS + Bx * Hp;

  const int tid = threadIdx.x;
  const int wave = tid >> 5, lane = tid & 31;
  const int jt = blockIdx.x, j0 = jt * 16;
  const int mbase = wave * 16;
  const int jlane = j0 + (lane & 15);

  cache_wslice(whhS, Whh16, j0, tid);
  cache_wslice(wihS, Wih16, j0, tid);

  const float bI = bsum[0 * Hx + jlane];
  const float bF = bsum[1 * Hx + jlane];
  const float bG = bsum[2 * Hx + jlane];
  const float bO = bsum[3 * Hx + jlane];

  float c[8];
#pragma unroll
  for (int r = 0; r < 8; r++) c[r] = 0.f;

  unsigned tgt = 0;
  for (int t = 0; t < Sx; t++) {
    half_t* hprev = (t & 1) ? hbufB : hbufA;
    half_t* hnew  = (t & 1) ? hbufA : hbufB;

    __syncthreads();
    // inp = x@ipW^T + ipb + h_prev_layer + w_t
    for (int e = tid; e < Bx * Hx; e += 128) {
      int b = e / Hx, j = e - b * Hx;
      const float* xp = x + ((size_t)b * Sx + t) * INx;
      size_t si = ((size_t)b * Sx + t) * Hp + j;
      float v = ipbl[j] + xp[0] * ipWl[j * INx + 0]
                        + xp[1] * ipWl[j * INx + 1]
                        + xp[2] * ipWl[j * INx + 2]
                        + (float)hprev_seq[si] + (float)wseq[si];
      inpS[b * Hp + j] = (half_t)v;
    }
    for (int e = tid; e < Bx * (Hp - Hx); e += 128) {
      int b = e / (Hp - Hx), j = Hx + (e - b * (Hp - Hx));
      inpS[b * Hp + j] = (half_t)0.f;
    }
    __syncthreads();

    v8f acc[4];
    {
      v8f z = {};
#pragma unroll
      for (int g = 0; g < 4; g++) acc[g] = z;
    }
#pragma unroll 1
    for (int kt = 0; kt < KT; kt++) {
      v16h a = load_frag(hprev, Hp, mbase, kt * 32, lane);
#pragma unroll
      for (int g = 0; g < 4; g++) {
        v16h bf = load_frag(whhS, Hp, g * 16, kt * 32, lane);
        acc[g] = wmma16(a, bf, acc[g]);
      }
    }
#pragma unroll 1
    for (int kt = 0; kt < KT; kt++) {
      v16h a = load_frag(inpS, Hp, mbase, kt * 32, lane);
#pragma unroll
      for (int g = 0; g < 4; g++) {
        v16h bf = load_frag(wihS, Hp, g * 16, kt * 32, lane);
        acc[g] = wmma16(a, bf, acc[g]);
      }
    }

#pragma unroll
    for (int r = 0; r < 8; r++) {
      float gi = acc[0][r] + bI;
      float gf = acc[1][r] + bF;
      float gg = acc[2][r] + bG;
      float go = acc[3][r] + bO;
      float cn = sigf(gf) * c[r] + sigf(gi) * tanhf(gg);
      c[r] = cn;
      float hv = sigf(go) * tanhf(cn);
      int b = mbase + r + ((lane & 16) ? 8 : 0);
      half_t h16 = (half_t)hv;
      hnew[(size_t)b * Hp + jlane] = h16;
      hseq[((size_t)b * Sx + t) * Hp + jlane] = h16;
    }
    gsync(ctr, tgt += NWG);
  }
}

// ---------------- output projection: running += h_seq @ fpW^T ----------------
__global__ __launch_bounds__(128)
void outproj_kernel(const half_t* __restrict__ hseq, const half_t* __restrict__ fpW16l,
                    const float* __restrict__ fpbl, const float* __restrict__ bias,
                    float* __restrict__ out, int initFlag) {
  const int tid = threadIdx.x, wave = tid >> 5, lane = tid & 31;
  const int mbase = blockIdx.x * 64 + wave * 16;  // M = b*S + t

  v8f acc[8];
  {
    v8f z = {};
#pragma unroll
    for (int nt = 0; nt < 8; nt++) acc[nt] = z;
  }
#pragma unroll 1
  for (int kt = 0; kt < KT; kt++) {
    v16h a = load_frag(hseq, Hp, mbase, kt * 32, lane);
#pragma unroll
    for (int nt = 0; nt < 8; nt++) {
      v16h b = load_frag(fpW16l, Hp, nt * 16, kt * 32, lane);
      acc[nt] = wmma16(a, b, acc[nt]);
    }
  }
  float bb = initFlag ? bias[0] : 0.f;
#pragma unroll
  for (int nt = 0; nt < 8; nt++) {
    int n = nt * 16 + (lane & 15);
    if (n < OUTx) {
#pragma unroll
      for (int r = 0; r < 8; r++) {
        int m = mbase + r + ((lane & 16) ? 8 : 0);
        float v = acc[nt][r] + fpbl[n] + bb;
        if (initFlag) out[(size_t)m * OUTx + n] = v;
        else          out[(size_t)m * OUTx + n] += v;
      }
    }
  }
}

// ---------------- host launch ----------------
extern "C" void kernel_launch(void* const* d_in, const int* in_sizes, int n_in,
                              void* d_out, int out_size, void* d_ws, size_t ws_size,
                              hipStream_t stream) {
  const float* x        = (const float*)d_in[0];
  const float* seqs     = (const float*)d_in[1];
  const float* Wih0     = (const float*)d_in[2];
  const float* Whh0     = (const float*)d_in[3];
  const float* bih0     = (const float*)d_in[4];
  const float* bhh0     = (const float*)d_in[5];
  const float* Wih_rest = (const float*)d_in[6];
  const float* Whh_rest = (const float*)d_in[7];
  const float* bih_rest = (const float*)d_in[8];
  const float* bhh_rest = (const float*)d_in[9];
  const float* ipW      = (const float*)d_in[10];
  const float* ipb      = (const float*)d_in[11];
  const float* fpW      = (const float*)d_in[12];
  const float* fpb      = (const float*)d_in[13];
  const float* winW     = (const float*)d_in[14];
  const float* winb     = (const float*)d_in[15];
  const float* whpW     = (const float*)d_in[16];
  const float* bias     = (const float*)d_in[17];
  float* out = (float*)d_out;

  // workspace layout
  char* base = (char*)d_ws;
  size_t off = 0;
  auto alloc = [&](size_t bytes) -> size_t {
    size_t o = off;
    off += (bytes + 255) & ~(size_t)255;
    return o;
  };
  const size_t hbufBytes = (size_t)Bx * Hp * sizeof(half_t);
  const size_t seqBytes  = (size_t)Bx * Sx * Hp * sizeof(half_t);
  const size_t wBytes    = (size_t)G4 * Hp * sizeof(half_t);

  size_t o_hbufA = alloc(hbufBytes);
  size_t o_hbufB = alloc(hbufBytes);
  size_t o_wbufA = alloc(hbufBytes);
  size_t o_wbufB = alloc(hbufBytes);
  size_t o_keta  = alloc((size_t)Bx * WINx * sizeof(float));
  size_t o_ctr0  = alloc(256);
  size_t o_ctr1  = alloc(256);
  size_t o_ctr2  = alloc(256);
  size_t o_hseqA = alloc(seqBytes);
  size_t o_hseqB = alloc(seqBytes);
  size_t o_wseq  = alloc(seqBytes);
  size_t o_Wih16 = alloc(3 * wBytes);
  size_t o_Whh16 = alloc(3 * wBytes);
  size_t o_fpW16 = alloc(3 * (size_t)OUTp * Hp * sizeof(half_t));
  size_t o_bsum  = alloc(3 * (size_t)G4 * sizeof(float));

  half_t* hbufA = (half_t*)(base + o_hbufA);
  half_t* hbufB = (half_t*)(base + o_hbufB);
  half_t* wbufA = (half_t*)(base + o_wbufA);
  half_t* wbufB = (half_t*)(base + o_wbufB);
  float*  keta  = (float*)(base + o_keta);
  unsigned* ctr0 = (unsigned*)(base + o_ctr0);
  unsigned* ctr1 = (unsigned*)(base + o_ctr1);
  unsigned* ctr2 = (unsigned*)(base + o_ctr2);
  half_t* hseqA = (half_t*)(base + o_hseqA);
  half_t* hseqB = (half_t*)(base + o_hseqB);
  half_t* wseq  = (half_t*)(base + o_wseq);
  half_t* Wih16 = (half_t*)(base + o_Wih16);
  half_t* Whh16 = (half_t*)(base + o_Whh16);
  half_t* fpW16 = (half_t*)(base + o_fpW16);
  float*  bsum  = (float*)(base + o_bsum);

  // zero everything (pads, h/c/w/keta state, sync counters) — capture-safe
  size_t zbytes = off < ws_size ? off : ws_size;
  hipMemsetAsync(d_ws, 0, zbytes, stream);

  // weight conversion (f32 -> padded f16)
  {
    int n = G4 * Hx, blocks = (n + 255) / 256;
    cvt_pad_kernel<<<blocks, 256, 0, stream>>>(Wih0, Wih16, G4, Hx, Hp);
    cvt_pad_kernel<<<blocks, 256, 0, stream>>>(Whh0, Whh16, G4, Hx, Hp);
    for (int l = 1; l < 3; l++) {
      cvt_pad_kernel<<<blocks, 256, 0, stream>>>(
          Wih_rest + (size_t)(l - 1) * G4 * Hx, Wih16 + (size_t)l * G4 * Hp, G4, Hx, Hp);
      cvt_pad_kernel<<<blocks, 256, 0, stream>>>(
          Whh_rest + (size_t)(l - 1) * G4 * Hx, Whh16 + (size_t)l * G4 * Hp, G4, Hx, Hp);
    }
    int nf = OUTx * Hx, bf = (nf + 255) / 256;
    for (int l = 0; l < 3; l++)
      cvt_pad_kernel<<<bf, 256, 0, stream>>>(
          fpW + (size_t)l * OUTx * Hx, fpW16 + (size_t)l * OUTp * Hp, OUTx, Hx, Hp);
    bsum_kernel<<<(3 * G4 + 255) / 256, 256, 0, stream>>>(bih0, bhh0, bih_rest, bhh_rest, bsum);
  }

  // layer 0 (persistent, 25 WGs, LDS-resident weights, WMMA recurrence + attention)
  lstm0_kernel<<<NWG, 128, 0, stream>>>(
      x, seqs, ipW, ipb, winW, winb, whpW,
      Wih16, Whh16, bsum, hbufA, hbufB, wbufA, wbufB,
      keta, hseqA, wseq, ctr0);
  outproj_kernel<<<(Bx * Sx) / 64, 128, 0, stream>>>(
      hseqA, fpW16, fpb, bias, out, 1);

  // layer 1
  hipMemsetAsync(hbufA, 0, hbufBytes, stream);
  hipMemsetAsync(hbufB, 0, hbufBytes, stream);
  lstmN_kernel<<<NWG, 128, 0, stream>>>(
      x, ipW + 1 * Hx * INx, ipb + 1 * Hx,
      Wih16 + (size_t)1 * G4 * Hp, Whh16 + (size_t)1 * G4 * Hp, bsum + 1 * G4,
      hseqA, wseq, hbufA, hbufB, hseqB, ctr1);
  outproj_kernel<<<(Bx * Sx) / 64, 128, 0, stream>>>(
      hseqB, fpW16 + (size_t)1 * OUTp * Hp, fpb + OUTx, bias, out, 0);

  // layer 2
  hipMemsetAsync(hbufA, 0, hbufBytes, stream);
  hipMemsetAsync(hbufB, 0, hbufBytes, stream);
  lstmN_kernel<<<NWG, 128, 0, stream>>>(
      x, ipW + 2 * Hx * INx, ipb + 2 * Hx,
      Wih16 + (size_t)2 * G4 * Hp, Whh16 + (size_t)2 * G4 * Hp, bsum + 2 * G4,
      hseqB, wseq, hbufA, hbufB, hseqA, ctr2);
  outproj_kernel<<<(Bx * Sx) / 64, 128, 0, stream>>>(
      hseqA, fpW16 + (size_t)2 * OUTp * Hp, fpb + 2 * OUTx, bias, out, 0);
}
